// ElmanRNN_85091892068660
// MI455X (gfx1250) — matrix operand order repaired and checked by reading, not compiled, not run on hardware
//
#include <hip/hip_runtime.h>

typedef __attribute__((ext_vector_type(16))) __bf16 v16bf;
typedef __attribute__((ext_vector_type(8)))  __bf16 v8bf;
typedef __attribute__((ext_vector_type(8)))  float  v8f;

#define NB 64      // batch
#define NT 1024    // time
#define NE 256     // embed dim
#define NH 512     // hidden dim
#define NO 32000   // output vocab
#define NK 768     // H + E

static __device__ __forceinline__ v16bf cat16(v8bf lo, v8bf hi) {
  return __builtin_shufflevector(lo, hi, 0,1,2,3,4,5,6,7,8,9,10,11,12,13,14,15);
}
// XOR-swizzled LDS index for the [64][512] bf16 hidden buffer.
// 16-byte granules; swizzle low 4 granule bits with row so the
// row-stride-1024B fragment loads spread across all 64 LDS banks.
static __device__ __forceinline__ int swzg(int g, int r) {
  return (g & 48) | ((g ^ r) & 15);
}
static __device__ __forceinline__ int swz(int r, int c) {
  return (r << 9) + (swzg(c >> 3, r) << 3) + (c & 7);
}
static __device__ __forceinline__ float tanh_fast(float x) {
  float cx = fminf(fmaxf(x, -12.0f), 12.0f);
  float e  = __expf(2.0f * cx);                        // v_exp_f32
  return (e - 1.0f) * __builtin_amdgcn_rcpf(e + 1.0f); // v_rcp_f32
}
static __device__ __forceinline__ v8bf cvt8(float4 a, float4 b) {
  v8bf v;
  v[0] = (__bf16)a.x; v[1] = (__bf16)a.y; v[2] = (__bf16)a.z; v[3] = (__bf16)a.w;
  v[4] = (__bf16)b.x; v[5] = (__bf16)b.y; v[6] = (__bf16)b.z; v[7] = (__bf16)b.w;
  return v;
}

// ---- Phase A: W_i2h f32 -> bf16, WMMA-B-fragment-blocked -----------------
// Layout: [ntile 0..31][kslab 0..23][lane 0..31][16 bf16]
// B frag element (lane,h) <-> W[nt*16 + (lane&15)][ks*32 + 8*(lane>>4) + (h&7) + 16*(h>>3)]
__global__ void wcvt_kernel(const float* __restrict__ w, __bf16* __restrict__ o) {
  int tid = blockIdx.x * 256 + threadIdx.x;   // 49152 threads, 8 elems each
  int run = tid & 1;
  int l   = (tid >> 1) & 31;
  int f   = tid >> 6;                         // fragment id = nt*24 + ks
  int ks  = f % 24, nt = f / 24;
  int row = nt * 16 + (l & 15);
  int col = ks * 32 + (l >> 4) * 8 + run * 16;
  const float4* src = (const float4*)(w + (size_t)row * NK + col);
  ((v8bf*)o)[tid] = cvt8(src[0], src[1]);     // dst elem index == tid*8
}

// ---- Phase B: embedding gather + cvt, WMMA-A-fragment-blocked ------------
// Layout: [t][mtile 0..3][kslab 0..7][lane 0..31][16 bf16]
// A frag runs: h=0..7 -> col k0+8*(lane>>4)+h ; h=8..15 -> col k0+16+8*(lane>>4)+(h-8)
__global__ void embed_kernel(const int* __restrict__ x, const float* __restrict__ tab,
                             __bf16* __restrict__ emb) {
  int tid = blockIdx.x * 256 + threadIdx.x;   // 2,097,152 threads, 8 elems each
  int run = tid & 1;
  int l   = (tid >> 1) & 31;
  int ks  = (tid >> 6) & 7;
  int mt  = (tid >> 9) & 3;
  int t   = tid >> 11;
  int b   = mt * 16 + (l & 15);
  int idx = x[b * NT + t];
  v8bf o = {};
  if (idx != 0) {                             // padding_idx=0 -> zeros
    int col = ks * 32 + (l >> 4) * 8 + run * 16;
    const float4* src = (const float4*)(tab + (size_t)idx * NE + col);
    o = cvt8(src[0], src[1]);
  }
  ((v8bf*)emb)[tid] = o;                      // dst elem index == tid*8
}

// ---- Phase C: persistent single-WGP recurrence ---------------------------
// 1024 threads = 32 waves = the 32 n-tiles of the [64x512] step output.
// 4 C tiles/wave (32 VGPRs). Weights re-streamed from L2 each step (786KB,
// fragment-blocked, fully coalesced); anti-hoist clobber keeps them out of
// registers across steps. Hidden state: swizzled 64KB LDS, 2 barriers/step.
__global__ void __launch_bounds__(1024)
rnn_kernel(const float* __restrict__ hidden0, const float* __restrict__ b_i2h,
           const __bf16* __restrict__ Wblk,   // blocked weights (phase A)
           const __bf16* __restrict__ Eblk,   // blocked embeddings (phase B)
           __bf16* __restrict__ hid_bf,       // [64][512] bf16 out (for decode)
           float*  __restrict__ hid_f32)      // d_out + B*O
{
  __shared__ __bf16 hidLds[NB * NH];          // 65536 bytes

  const int tid  = threadIdx.x;
  const int lane = tid & 31;
  const int wv   = tid >> 5;                  // 0..31 = n-tile
  const int l15  = lane & 15;
  const int hi   = lane >> 4;

  for (int i = tid; i < NB * NH; i += 1024) {
    int r = i >> 9, c = i & 511;
    hidLds[swz(r, c)] = (__bf16)hidden0[i];
  }
  __syncthreads();

  const int   ncol  = wv * 16 + l15;
  const float bias  = b_i2h[ncol];
  const __bf16* wfrag = Wblk + ((size_t)wv * 24 * 32 + lane) * 16;  // + ks*512
  const __bf16* efragL = Eblk + (size_t)lane * 16;                  // + frag*512

  for (int t = 0; t < NT; ++t) {
    asm volatile("" ::: "memory");            // forbid cross-step load hoisting
    const __bf16* efrag = efragL + (size_t)t * (4 * 8 * 512);
    v8f acc[4];
#pragma unroll
    for (int mt = 0; mt < 4; ++mt) acc[mt] = (v8f){};

    // K slabs 0..15: hidden part of A (LDS, swizzled)
    for (int ks = 0; ks < 16; ++ks) {
      const __bf16* wk = wfrag + ks * 512;
      v16bf bf = cat16(*(const v8bf*)wk, *(const v8bf*)(wk + 8));
      const int g1 = ks * 4 + hi, g2 = g1 + 2;
      v16bf af[4];
#pragma unroll
      for (int mt = 0; mt < 4; ++mt) {
        const int r = mt * 16 + l15;
        af[mt] = cat16(*(const v8bf*)&hidLds[(r << 9) + (swzg(g1, r) << 3)],
                       *(const v8bf*)&hidLds[(r << 9) + (swzg(g2, r) << 3)]);
      }
#pragma unroll
      for (int mt = 0; mt < 4; ++mt)
        acc[mt] = __builtin_amdgcn_wmma_f32_16x16x32_bf16(
            false, af[mt], false, bf, (short)0, acc[mt], false, false);
    }
    // K slabs 16..23: embedding part of A (global, fragment-blocked, L2-hot)
    for (int ks = 0; ks < 8; ++ks) {
      const __bf16* wk = wfrag + (16 + ks) * 512;
      v16bf bf = cat16(*(const v8bf*)wk, *(const v8bf*)(wk + 8));
      v16bf af[4];
#pragma unroll
      for (int mt = 0; mt < 4; ++mt) {
        const __bf16* ea = efrag + (mt * 8 + ks) * 512;
        af[mt] = cat16(*(const v8bf*)ea, *(const v8bf*)(ea + 8));
      }
#pragma unroll
      for (int mt = 0; mt < 4; ++mt)
        acc[mt] = __builtin_amdgcn_wmma_f32_16x16x32_bf16(
            false, af[mt], false, bf, (short)0, acc[mt], false, false);
    }

    __syncthreads();   // all waves done READING h_{t-1}
#pragma unroll
    for (int mt = 0; mt < 4; ++mt)
#pragma unroll
      for (int r8 = 0; r8 < 8; ++r8) {
        const int row = mt * 16 + r8 + hi * 8;        // C layout: M = r + 8*hiHalf
        hidLds[swz(row, ncol)] = (__bf16)tanh_fast(acc[mt][r8] + bias);
      }
    __syncthreads();   // h_t visible before next step's reads
  }

  for (int i = tid; i < NB * NH; i += 1024) {
    int r = i >> 9, c = i & 511;
    __bf16 v = hidLds[swz(r, c)];
    hid_bf[i]  = v;
    hid_f32[i] = (float)v;
  }
}

// ---- Phase D: decode GEMM [64x512] x [512x32000] -------------------------
// Bandwidth-bound (65MB W_dec read once). Wave per 16x16 tile; f32->bf16
// conversion in-register for B fragments.
__global__ void __launch_bounds__(128)
decode_kernel(const __bf16* __restrict__ hid,   // [64][512] bf16 row-major
              const float* __restrict__ Wdec,   // [32000][512] f32
              const float* __restrict__ bdec,
              float* __restrict__ out)          // [64][32000]
{
  const int gw   = (blockIdx.x * 128 + threadIdx.x) >> 5;  // 0..7999
  const int lane = threadIdx.x & 31;
  const int l15  = lane & 15;
  const int hi   = lane >> 4;
  const int mt   = gw & 3;
  const int nt   = gw >> 2;

  const float* wrow = Wdec + (size_t)(nt * 16 + l15) * NH;
  v8f c = {};
  for (int ks = 0; ks < 16; ++ks) {
    const int k0 = ks * 32;
    const __bf16* ar = hid + (size_t)(mt * 16 + l15) * NH + k0 + hi * 16;
    v16bf a = cat16(*(const v8bf*)ar, *(const v8bf*)(ar + 8));

    const float* wk = wrow + k0 + hi * 16;
    v16bf b = cat16(cvt8(((const float4*)wk)[0], ((const float4*)wk)[1]),
                    cvt8(((const float4*)wk)[2], ((const float4*)wk)[3]));
    c = __builtin_amdgcn_wmma_f32_16x16x32_bf16(false, a, false, b, (short)0, c,
                                                false, false);
  }
  const float bias = bdec[nt * 16 + l15];
#pragma unroll
  for (int r = 0; r < 8; ++r)
    out[(size_t)(mt * 16 + r + hi * 8) * NO + nt * 16 + l15] = c[r] + bias;
}

extern "C" void kernel_launch(void* const* d_in, const int* in_sizes, int n_in,
                              void* d_out, int out_size, void* d_ws, size_t ws_size,
                              hipStream_t stream) {
  (void)in_sizes; (void)n_in; (void)out_size; (void)ws_size;
  const int*   x      = (const int*)d_in[0];
  const float* hidden = (const float*)d_in[1];
  const float* tab    = (const float*)d_in[2];
  const float* Wi2h   = (const float*)d_in[3];
  const float* bi2h   = (const float*)d_in[4];
  const float* Wdec   = (const float*)d_in[5];
  const float* bdec   = (const float*)d_in[6];
  float* out = (float*)d_out;

  char* ws = (char*)d_ws;
  __bf16* wblk   = (__bf16*)ws;                            // 786,432 B
  __bf16* eblk   = (__bf16*)(ws + 786432);                 // 33,554,432 B
  __bf16* hid_bf = (__bf16*)(ws + 786432 + 33554432);      // 65,536 B

  wcvt_kernel<<<192, 256, 0, stream>>>(Wi2h, wblk);              // 49,152 thr
  embed_kernel<<<8192, 256, 0, stream>>>(x, tab, eblk);          // 2,097,152 thr
  rnn_kernel<<<1, 1024, 0, stream>>>(hidden, bi2h, wblk, eblk,
                                     hid_bf, out + (size_t)NB * NO);
  decode_kernel<<<2000, 128, 0, stream>>>(hid_bf, Wdec, bdec, out);
}
